// LSTM_70892730188190
// MI455X (gfx1250) — matrix operand order repaired and checked by reading, not compiled, not run on hardware
//
#include <hip/hip_runtime.h>

// ---------------------------------------------------------------------------
// 2-layer LSTM forward on MI455X (gfx1250) — PERSISTENT bf16 WMMA kernel.
//   gates[B,4H] = concat(x,h)[B,2048] @ W^T[2048,4096] + b   per (t, layer)
// - ONE launch runs all 512 timesteps x 2 layers; a device-wide atomic
//   barrier (release add + acquire spin + s_sleep) orders the 16 blocks
//   between phases, eliminating 1024 sequential kernel launches.
// - Activation slab (32 x 2048 bf16) staged into LDS via Tensor Data Mover
//   each phase (wave 0 issues tensor_load_to_lds, s_wait_tensorcnt).
// - LDS rows padded (+4 dwords / 256 dwords via D# pad feature) to stagger
//   banks while keeping 16B alignment for ds_load_b128.
// - Each wave owns one 16-col N-tile and BOTH 16-row M-tiles for all four
//   gates (8 f32 accumulators): every weight tile feeds two WMMAs, so W is
//   streamed exactly once per step.
// - Weight tiles and LDS A-tiles double-buffered in registers so the 8 WMMAs
//   overlap the next iteration's global + LDS loads.
// ---------------------------------------------------------------------------

typedef __attribute__((ext_vector_type(16))) __bf16 v16bf;
typedef __attribute__((ext_vector_type(8)))  __bf16 v8bf;
typedef __attribute__((ext_vector_type(8)))  float  v8f;
typedef unsigned int u32x4 __attribute__((ext_vector_type(4)));
typedef int          i32x4 __attribute__((ext_vector_type(4)));
typedef int          i32x8 __attribute__((ext_vector_type(8)));

constexpr int BATCH = 32;
constexpr int SEQ   = 512;
constexpr int IN    = 1024;
constexpr int HID   = 1024;
constexpr int KTOT  = IN + HID;   // 2048
constexpr int G4    = 4 * HID;    // 4096
constexpr int NBLK  = 16;         // persistent blocks (device barrier members)

// LDS slab: 32 rows x 1024 bf16; 512 dw/row + 4 pad dw per 256 dw -> 520 dw/row
constexpr int ROW_DW      = 520;
constexpr int SLAB_USHORT = (32 * ROW_DW + 32) * 2;   // margin for trailing pad

// ---- helpers ---------------------------------------------------------------

__device__ __forceinline__ unsigned short f2bf(float f) {
  unsigned int u = __float_as_uint(f);
  u += 0x7FFFu + ((u >> 16) & 1u);   // round-to-nearest-even
  return (unsigned short)(u >> 16);
}

__device__ __forceinline__ v8bf ld8(const unsigned short* p) {
  return *reinterpret_cast<const v8bf*>(p);   // 16-byte global load
}

// padded LDS read: rowbase = row*ROW_DW (dwords), colElem in bf16 elements
__device__ __forceinline__ v8bf lds_at(const unsigned short* slab, int rowbase, int colElem) {
  int cd = colElem >> 1;                       // row-local dword column
  int ad = rowbase + cd + ((cd >> 8) << 2);    // +4 dwords per 256 (TDM pad)
  return *reinterpret_cast<const v8bf*>(slab + (ad << 1));
}

__device__ __forceinline__ float sigf(float x) {
  return __frcp_rn(1.0f + __expf(-x));
}
__device__ __forceinline__ float tanhfast(float x) {
  return 2.0f * __frcp_rn(1.0f + __expf(-2.0f * x)) - 1.0f;
}

// ---- Tensor Data Mover: 2D tile (rows x cols bf16) -> LDS ------------------
// pad: 4 dwords every 256 dwords (pad_interval=7 -> 256dw, pad_amount=3 -> 4dw)
__device__ __forceinline__ void tdm_load_2d(unsigned lds_off, const void* gptr,
                                            unsigned rows, unsigned cols,
                                            unsigned long long row_stride_elems) {
  unsigned long long ga = (unsigned long long)(uintptr_t)gptr;
  u32x4 g0;
  g0[0] = 1u;                                            // count=1, user desc
  g0[1] = lds_off;                                       // lds_addr [63:32]
  g0[2] = (unsigned)ga;                                  // global_addr low
  g0[3] = (unsigned)((ga >> 32) & 0x01FFFFFFu)           // global_addr[56:32]
        | 0x80000000u;                                   // type=2 ("image")
  i32x8 g1;
  g1[0] = (int)((1u << 16)        // data_size = 2 bytes
              | (1u << 20)        // pad_enable
              | (7u << 22)        // pad_interval: 256 dwords
              | (3u << 25));      // pad_amount: 4 dwords
  g1[1] = (int)((cols & 0xFFFFu) << 16);                 // tensor_dim0 [15:0]
  g1[2] = (int)(((cols >> 16) & 0xFFFFu) | ((rows & 0xFFFFu) << 16)); // dim0 hi | dim1 lo
  g1[3] = (int)((cols & 0xFFFFu) << 16);                 // tile_dim0
  g1[4] = (int)(rows & 0xFFFFu);                         // tile_dim1 (tile_dim2=0)
  g1[5] = (int)(unsigned)(row_stride_elems & 0xFFFFFFFFu); // tensor_dim0_stride lo
  g1[6] = (int)(unsigned)((row_stride_elems >> 32) & 0xFFFFu);
  g1[7] = 0;
  i32x4 z4 = {0, 0, 0, 0};
#if defined(__clang_major__) && (__clang_major__ >= 23)
  i32x8 z8 = {0, 0, 0, 0, 0, 0, 0, 0};
  __builtin_amdgcn_tensor_load_to_lds(g0, g1, z4, z4, z8, 0);
#else
  __builtin_amdgcn_tensor_load_to_lds(g0, g1, z4, z4, 0);
#endif
}

// ---- device-wide barrier over NBLK blocks (monotonic counter) --------------
__device__ __forceinline__ void grid_barrier(unsigned* ctr, unsigned target) {
  __syncthreads();
  if (threadIdx.x == 0) {
    __hip_atomic_fetch_add(ctr, 1u, __ATOMIC_RELEASE, __HIP_MEMORY_SCOPE_AGENT);
    while (__hip_atomic_load(ctr, __ATOMIC_ACQUIRE, __HIP_MEMORY_SCOPE_AGENT) < target)
      __builtin_amdgcn_s_sleep(2);
  }
  __syncthreads();
}

// ---- fp32 -> bf16 conversion (weights / input, once per call) --------------

__global__ void cvt_f32_bf16(const float* __restrict__ src,
                             unsigned short* __restrict__ dst, int n) {
  int i = blockIdx.x * blockDim.x + threadIdx.x;
  int stride = gridDim.x * blockDim.x;
  for (; i < n; i += stride) dst[i] = f2bf(src[i]);
}

// ---- one (t, layer) phase: GEMM(4 gates x 2 M-tiles) + LSTM cell -----------
// 64 waves total (16 blocks x 4 waves), wave = one 16-col N-tile, M=32.

__device__ __forceinline__ void step_body(
    unsigned short* shX, unsigned short* shH,
    const unsigned short* __restrict__ Ax, int ldAx,
    const unsigned short* __restrict__ Ah,
    const unsigned short* __restrict__ Wb,
    const float*          __restrict__ bias,
    float*                __restrict__ Cst,
    unsigned short*       __restrict__ Hout,
    float*                __restrict__ Out, int t,
    float*                __restrict__ HnOut,
    float*                __restrict__ CnOut)
{
  const int lane    = threadIdx.x & 31;
  const int wid     = (blockIdx.x << 2) | (threadIdx.x >> 5);  // 0..63
  const int n0      = wid << 4;                                // 0..1008
  const int laneCol = lane & 15;
  const int laneHi  = lane >> 4;

  // ---- wave 0: TDM-load both activation slabs (all 32 batch rows) ----
  if ((threadIdx.x >> 5) == 0) {
    tdm_load_2d((unsigned)(uintptr_t)shX, Ax, 32u, (unsigned)IN,
                (unsigned long long)ldAx);
    tdm_load_2d((unsigned)(uintptr_t)shH, Ah, 32u, (unsigned)HID,
                (unsigned long long)HID);
    __builtin_amdgcn_s_wait_tensorcnt(0);
  }
  __syncthreads();

  v8f acc[2][4];
  #pragma unroll
  for (int mt = 0; mt < 2; ++mt)
    #pragma unroll
    for (int gi = 0; gi < 4; ++gi)
      acc[mt][gi] = v8f{0.f, 0.f, 0.f, 0.f, 0.f, 0.f, 0.f, 0.f};

  // A-slab row bases (dwords): M-tile 0 -> rows 0..15, M-tile 1 -> rows 16..31
  const int rb0 = laneCol * ROW_DW;
  const int rb1 = (16 + laneCol) * ROW_DW;

  // W^T column n == W row n (contiguous in k); per-lane base
  const unsigned short* wcol = Wb + (size_t)(n0 + laneCol) * KTOT + laneHi * 8;
  constexpr size_t GOFF = (size_t)HID * KTOT;   // per-gate row-block offset

  // ---- preload kc = 0 (register double buffers for B and A) ----
  v8bf blo[4], bhi[4];
  #pragma unroll
  for (int gi = 0; gi < 4; ++gi) {
    blo[gi] = ld8(wcol + (size_t)gi * GOFF);
    bhi[gi] = ld8(wcol + (size_t)gi * GOFF + 16);
  }
  v8bf a0lo = lds_at(shX, rb0, laneHi * 8);
  v8bf a0hi = lds_at(shX, rb0, 16 + laneHi * 8);
  v8bf a1lo = lds_at(shX, rb1, laneHi * 8);
  v8bf a1hi = lds_at(shX, rb1, 16 + laneHi * 8);

  #pragma unroll 2
  for (int kc = 0; kc < KTOT; kc += 32) {
    const int kn = (kc + 32 < KTOT) ? kc + 32 : kc;

    // issue next iteration's weight loads
    v8bf nblo[4], nbhi[4];
    #pragma unroll
    for (int gi = 0; gi < 4; ++gi) {
      const unsigned short* wp = wcol + (size_t)gi * GOFF + kn;
      __builtin_prefetch(wp + 64, 0, 1);     // next-next slab -> global_prefetch_b8
      nblo[gi] = ld8(wp);
      nbhi[gi] = ld8(wp + 16);
    }
    // issue next iteration's A-tile LDS loads
    const unsigned short* nslab = (kn < IN) ? shX : shH;
    const int nkl = (kn < IN) ? kn : kn - IN;
    v8bf na0lo = lds_at(nslab, rb0, nkl + laneHi * 8);
    v8bf na0hi = lds_at(nslab, rb0, nkl + 16 + laneHi * 8);
    v8bf na1lo = lds_at(nslab, rb1, nkl + laneHi * 8);
    v8bf na1hi = lds_at(nslab, rb1, nkl + 16 + laneHi * 8);

    // 8 WMMAs on current tiles: each B-tile feeds both M-tiles
    v16bf a0 = __builtin_shufflevector(a0lo, a0hi, 0,1,2,3,4,5,6,7,8,9,10,11,12,13,14,15);
    v16bf a1 = __builtin_shufflevector(a1lo, a1hi, 0,1,2,3,4,5,6,7,8,9,10,11,12,13,14,15);
    #pragma unroll
    for (int gi = 0; gi < 4; ++gi) {
      v16bf b = __builtin_shufflevector(blo[gi], bhi[gi],
                                        0,1,2,3,4,5,6,7,8,9,10,11,12,13,14,15);
      acc[0][gi] = __builtin_amdgcn_wmma_f32_16x16x32_bf16(
          false, a0, false, b, (short)0, acc[0][gi], false, false);
      acc[1][gi] = __builtin_amdgcn_wmma_f32_16x16x32_bf16(
          false, a1, false, b, (short)0, acc[1][gi], false, false);
    }
    // rotate double buffers
    #pragma unroll
    for (int gi = 0; gi < 4; ++gi) { blo[gi] = nblo[gi]; bhi[gi] = nbhi[gi]; }
    a0lo = na0lo; a0hi = na0hi; a1lo = na1lo; a1hi = na1hi;
  }

  // ---- bias + cell update (all accs share the C/D VGPR layout) ----
  const int n   = n0 + laneCol;
  const float bF = bias[0 * HID + n];
  const float bI = bias[1 * HID + n];
  const float bG = bias[2 * HID + n];
  const float bO = bias[3 * HID + n];

  #pragma unroll
  for (int mt = 0; mt < 2; ++mt) {
    const int mRow = mt * 16 + laneHi * 8;   // VGPR r, lane l -> M = r + 8*(l>=16)
    #pragma unroll
    for (int r = 0; r < 8; ++r) {
      const int m = mRow + r;
      const size_t ci = (size_t)m * HID + n;
      float f = sigf    (acc[mt][0][r] + bF);
      float i = sigf    (acc[mt][1][r] + bI);
      float g = tanhfast(acc[mt][2][r] + bG);
      float o = sigf    (acc[mt][3][r] + bO);
      float cn = f * Cst[ci] + i * g;
      float h  = o * tanhfast(cn);
      Cst[ci]  = cn;
      Hout[ci] = f2bf(h);
      if (Out)   Out[(size_t)m * (SEQ * HID) + (size_t)t * HID + n] = h;
      if (HnOut) { HnOut[ci] = h; CnOut[ci] = cn; }
    }
  }
}

// ---- persistent kernel: all 512 timesteps x 2 layers -----------------------

__global__ __launch_bounds__(128, 1)
void lstm_persistent(const unsigned short* __restrict__ Xbf,
                     const unsigned short* __restrict__ Wb0,
                     const float*          __restrict__ b0,
                     const unsigned short* __restrict__ Wb1,
                     const float*          __restrict__ b1,
                     unsigned short*       __restrict__ Hbf,   // [layer][buf] 4 x BH
                     float*                __restrict__ Cst,   // [layer] 2 x BH
                     float*                __restrict__ out,   // outputs | h_n | c_n
                     unsigned*             __restrict__ bar)   // zeroed barrier ctr
{
  __shared__ __align__(16) unsigned short shX[SLAB_USHORT];  // x-part 32x1024
  __shared__ __align__(16) unsigned short shH[SLAB_USHORT];  // h-part 32x1024

  const size_t BH  = (size_t)BATCH * HID;
  const size_t OUT = (size_t)BATCH * SEQ * HID;
  unsigned phase = 0;

  for (int t = 0; t < SEQ; ++t) {
    const int rb = t & 1, wb = rb ^ 1;
    unsigned short* h0r = Hbf + (size_t)(0 + rb) * BH;
    unsigned short* h0w = Hbf + (size_t)(0 + wb) * BH;
    unsigned short* h1r = Hbf + (size_t)(2 + rb) * BH;
    unsigned short* h1w = Hbf + (size_t)(2 + wb) * BH;
    const bool last = (t == SEQ - 1);

    // layer 0: x-part from converted input x[:, t, :] (row stride SEQ*IN)
    step_body(shX, shH,
              Xbf + (size_t)t * IN, SEQ * IN, h0r, Wb0, b0,
              Cst, h0w, /*Out=*/nullptr, t,
              last ? out + OUT + 0 * BH : nullptr,
              last ? out + OUT + 2 * BH : nullptr);
    grid_barrier(bar, ++phase * NBLK);

    // layer 1: x-part is layer-0 h at current t (row stride HID)
    step_body(shX, shH,
              h0w, HID, h1r, Wb1, b1,
              Cst + BH, h1w, /*Out=*/out, t,
              last ? out + OUT + 1 * BH : nullptr,
              last ? out + OUT + 3 * BH : nullptr);
    grid_barrier(bar, ++phase * NBLK);
  }
}

// ---------------------------------------------------------------------------

extern "C" void kernel_launch(void* const* d_in, const int* in_sizes, int n_in,
                              void* d_out, int out_size, void* d_ws, size_t ws_size,
                              hipStream_t stream) {
  const float* x  = (const float*)d_in[0];
  const float* W0 = (const float*)d_in[1];
  const float* b0 = (const float*)d_in[2];
  const float* W1 = (const float*)d_in[3];
  const float* b1 = (const float*)d_in[4];
  float* out = (float*)d_out;

  // workspace layout (bf16 buffers first, then fp32 cell state, then barrier)
  const size_t WSZ = (size_t)G4 * KTOT;          // 8,388,608 elems
  const size_t XSZ = (size_t)BATCH * SEQ * IN;   // 16,777,216 elems
  const size_t BH  = (size_t)BATCH * HID;        // 32,768 elems
  unsigned short* Wb0 = (unsigned short*)d_ws;
  unsigned short* Wb1 = Wb0 + WSZ;
  unsigned short* Xbf = Wb1 + WSZ;
  unsigned short* Hbf = Xbf + XSZ;               // [layer][buf] 4 x BH bf16
  float*          Cst = (float*)(Hbf + 4 * BH);  // [layer] 2 x BH f32
  unsigned*       Bar = (unsigned*)(Cst + 2 * BH);

  // zero h (bf16), c (f32) and the barrier counter — contiguous region
  hipMemsetAsync(Hbf, 0,
                 4 * BH * sizeof(unsigned short) + 2 * BH * sizeof(float) + 256,
                 stream);

  cvt_f32_bf16<<<2048, 256, 0, stream>>>(W0, Wb0, (int)WSZ);
  cvt_f32_bf16<<<2048, 256, 0, stream>>>(W1, Wb1, (int)WSZ);
  cvt_f32_bf16<<<4096, 256, 0, stream>>>(x,  Xbf, (int)XSZ);

  lstm_persistent<<<NBLK, 128, 0, stream>>>(Xbf, Wb0, b0, Wb1, b1,
                                            Hbf, Cst, out, Bar);
}
